// NeuralModel_80573586473051
// MI455X (gfx1250) — matrix operand order repaired and checked by reading, not compiled, not run on hardware
//
#include <hip/hip_runtime.h>
#include <math.h>

// ---------------- model dims ----------------
#define TT 512
#define DIM 512
#define HEADS 12
#define DH 64
#define INNER (HEADS * DH)      // 768
#define DEPTH 12
#define FF_INNER 2048
#define CF 4098                 // C * FBINS = 2 * 2049
#define NOUT 8196               // FBINS * OUT_MASKS = 2049 * 4
#define ALIBI_HEADS 4
#define SCALE 0.125f            // 64^-0.5
#define NEG_INF (-1.0e9f)

typedef __attribute__((ext_vector_type(16))) __bf16 v16bf;
typedef __attribute__((ext_vector_type(4)))  __bf16 v4bf;
typedef __attribute__((ext_vector_type(8)))  float v8f;

// ---------------- generic strided batched WMMA GEMM ----------------
// C[b][m][n] = scale * sum_k A[b][m][k] * B[b][k][n]  (+ bias[n]) (+ resid, same layout as C)
// Block tile 128x64, BK=32, 256 threads = 8 wave32s, each wave computes 32x32 (4 WMMAs/step).
// Software-pipelined: double-buffered LDS; global loads of tile t+1 issue before the WMMAs of
// tile t so the f32->bf16 convert+LDS store (which waits on loadcnt) lands after the compute.
// Template: AG = guard A staging (K ragged), BG = guard B staging + epilogue (K or N ragged),
//           AK1 = A k-contiguous (vector staging), BN1 = B n-contiguous (vector staging).
// M is always a multiple of 128 in this model (M = 512 everywhere).
#define BM 128
#define BN 64
#define BK 32
#define LPAD 8

template <bool AG, bool BG, bool AK1, bool BN1>
__global__ __launch_bounds__(256)
void wmma_gemm_kernel(const float* __restrict__ A, const float* __restrict__ B,
                      const float* __restrict__ bias, const float* __restrict__ resid,
                      float* __restrict__ C,
                      int M, int N, int K,
                      long As0, long As1, long Bs0, long Bs1, long Cs0, long Cs1,
                      long batA, long batB, long batC, float scale)
{
    __shared__ __bf16 sA[2][BM][BK + LPAD];
    __shared__ __bf16 sB[2][BN][BK + LPAD];   // stored transposed: sB[buf][n][k]

    const int tid  = threadIdx.x;
    const int lane = tid & 31;
    const int wave = tid >> 5;   // 0..7
    const int wm   = wave >> 1;  // 0..3  (row block of 32)
    const int wn   = wave & 1;   // 0..1  (col block of 32)
    const int l16  = lane & 15;
    const int halfl = lane >> 4;

    const int m0 = blockIdx.y * BM;
    const int n0 = blockIdx.x * BN;
    const int bz = blockIdx.z;

    const float* Ab = A + (long)bz * batA;
    const float* Bb = B + (long)bz * batB;
    float*       Cb = C + (long)bz * batC;
    const float* Rb = resid ? resid + (long)bz * batC : nullptr;

    v8f acc00 = {}, acc01 = {}, acc10 = {}, acc11 = {};

    // register staging for one tile
    float4 ra[4]; float va[16];
    float4 rb[2]; float vb[8];

    auto load_tiles = [&](int k0) {
        if (AK1 && !AG) {
            #pragma unroll
            for (int i = 0; i < 4; ++i) {
                int idx = tid + i * 256;          // 0..1023
                int m = idx >> 3;                 // 0..127
                int k4 = (idx & 7) << 2;          // 0,4,...,28
                ra[i] = *(const float4*)(Ab + (long)(m0 + m) * As0 + (k0 + k4));
            }
        } else {
            #pragma unroll
            for (int i = 0; i < 16; ++i) {
                int idx = tid + i * 256;          // 0..4095
                int m = idx >> 5, k = idx & 31;
                int gm = m0 + m, gk = k0 + k;
                if (AG) {
                    int cm = gm < M ? gm : M - 1;
                    int ck = gk < K ? gk : K - 1;
                    float v = Ab[(long)cm * As0 + (long)ck * As1];
                    va[i] = (gm < M && gk < K) ? v : 0.0f;
                } else {
                    va[i] = Ab[(long)gm * As0 + (long)gk * As1];
                }
            }
        }
        if (BN1 && !BG) {
            #pragma unroll
            for (int i = 0; i < 2; ++i) {
                int idx = tid + i * 256;          // 0..511
                int k = idx >> 4;                 // 0..31
                int n4 = (idx & 15) << 2;         // 0,4,...,60
                rb[i] = *(const float4*)(Bb + (long)(k0 + k) * Bs0 + (n0 + n4));
            }
        } else {
            #pragma unroll
            for (int i = 0; i < 8; ++i) {
                int idx = tid + i * 256;          // 0..2047
                int k = idx >> 6, n = idx & 63;
                int gk = k0 + k, gn = n0 + n;
                if (BG) {
                    int ck = gk < K ? gk : K - 1;
                    int cn = gn < N ? gn : N - 1;
                    float v = Bb[(long)ck * Bs0 + (long)cn * Bs1];
                    vb[i] = (gk < K && gn < N) ? v : 0.0f;
                } else {
                    vb[i] = Bb[(long)gk * Bs0 + (long)gn * Bs1];
                }
            }
        }
    };

    auto store_tiles = [&](int buf) {
        if (AK1 && !AG) {
            #pragma unroll
            for (int i = 0; i < 4; ++i) {
                int idx = tid + i * 256;
                int m = idx >> 3;
                int k4 = (idx & 7) << 2;
                v4bf bv;
                bv[0] = (__bf16)ra[i].x; bv[1] = (__bf16)ra[i].y;
                bv[2] = (__bf16)ra[i].z; bv[3] = (__bf16)ra[i].w;
                *(v4bf*)&sA[buf][m][k4] = bv;
            }
        } else {
            #pragma unroll
            for (int i = 0; i < 16; ++i) {
                int idx = tid + i * 256;
                int m = idx >> 5, k = idx & 31;
                sA[buf][m][k] = (__bf16)va[i];
            }
        }
        if (BN1 && !BG) {
            #pragma unroll
            for (int i = 0; i < 2; ++i) {
                int idx = tid + i * 256;
                int k = idx >> 4;
                int n4 = (idx & 15) << 2;
                sB[buf][n4 + 0][k] = (__bf16)rb[i].x;
                sB[buf][n4 + 1][k] = (__bf16)rb[i].y;
                sB[buf][n4 + 2][k] = (__bf16)rb[i].z;
                sB[buf][n4 + 3][k] = (__bf16)rb[i].w;
            }
        } else {
            #pragma unroll
            for (int i = 0; i < 8; ++i) {
                int idx = tid + i * 256;
                int k = idx >> 6, n = idx & 63;
                sB[buf][n][k] = (__bf16)vb[i];
            }
        }
    };

    auto compute = [&](int buf) {
        // fragments per ISA wave32 VGPR layouts
        v16bf a0, a1, b0, b1;
        int ma = wm * 32 + l16;
        #pragma unroll
        for (int r = 0; r < 8; ++r) {
            // A 16x32 bf16: lane holds row M=l16; VGPR r -> K = 16*(r/4) + 8*half + 2*(r%4)
            int k = ((r >> 2) << 4) + (halfl << 3) + ((r & 3) << 1);
            a0[2 * r]     = sA[buf][ma][k];
            a0[2 * r + 1] = sA[buf][ma][k + 1];
            a1[2 * r]     = sA[buf][ma + 16][k];
            a1[2 * r + 1] = sA[buf][ma + 16][k + 1];
        }
        int na = wn * 32 + l16;
        #pragma unroll
        for (int r = 0; r < 8; ++r) {
            // B 32x16 bf16: lane holds col N=l16; VGPR r -> K = 16*(r/4) + 8*half + 2*(r%4)
            int k = ((r >> 2) << 4) + (halfl << 3) + ((r & 3) << 1);
            b0[2 * r]     = sB[buf][na][k];
            b0[2 * r + 1] = sB[buf][na][k + 1];
            b1[2 * r]     = sB[buf][na + 16][k];
            b1[2 * r + 1] = sB[buf][na + 16][k + 1];
        }
        acc00 = __builtin_amdgcn_wmma_f32_16x16x32_bf16(false, a0, false, b0, (short)0, acc00, false, false);
        acc01 = __builtin_amdgcn_wmma_f32_16x16x32_bf16(false, a0, false, b1, (short)0, acc01, false, false);
        acc10 = __builtin_amdgcn_wmma_f32_16x16x32_bf16(false, a1, false, b0, (short)0, acc10, false, false);
        acc11 = __builtin_amdgcn_wmma_f32_16x16x32_bf16(false, a1, false, b1, (short)0, acc11, false, false);
    };

    const int nt = (K + BK - 1) / BK;
    load_tiles(0);
    store_tiles(0);
    __syncthreads();
    for (int t = 0; t < nt; ++t) {
        int cur = t & 1;
        if (t + 1 < nt) load_tiles((t + 1) * BK);   // in flight during compute
        compute(cur);
        if (t + 1 < nt) {
            store_tiles(cur ^ 1);                   // waits loadcnt here, after WMMAs
            __syncthreads();
        }
    }

    // ---- epilogue: C 16x16 f32 layout — VGPR r holds row 8*half + r, col = l16 ----
    const int ca = n0 + wn * 32 + l16;
    const int cb = ca + 16;
    float biasa = 0.0f, biasb = 0.0f;
    if (bias) {
        biasa = (!BG || ca < N) ? bias[ca] : 0.0f;
        biasb = (!BG || cb < N) ? bias[cb] : 0.0f;
    }
    #pragma unroll
    for (int r = 0; r < 8; ++r) {
        int m_a = m0 + wm * 32 + (halfl << 3) + r;        // rows of acc0x
        int m_b = m_a + 16;                               // rows of acc1x
        if (!BG || ca < N) {
            long offA = (long)m_a * Cs0 + (long)ca * Cs1;
            long offB = (long)m_b * Cs0 + (long)ca * Cs1;
            float v0 = acc00[r] * scale + biasa;
            float v1 = acc10[r] * scale + biasa;
            if (Rb) { v0 += Rb[offA]; v1 += Rb[offB]; }
            Cb[offA] = v0;
            Cb[offB] = v1;
        }
        if (!BG || cb < N) {
            long offA = (long)m_a * Cs0 + (long)cb * Cs1;
            long offB = (long)m_b * Cs0 + (long)cb * Cs1;
            float v0 = acc01[r] * scale + biasb;
            float v1 = acc11[r] * scale + biasb;
            if (Rb) { v0 += Rb[offA]; v1 += Rb[offB]; }
            Cb[offA] = v0;
            Cb[offB] = v1;
        }
    }
}

// ---------------- LayerNorm (row of 512, block = 256 threads) ----------------
__global__ __launch_bounds__(256)
void layernorm_kernel(const float* __restrict__ x, const float* __restrict__ g,
                      const float* __restrict__ b, float* __restrict__ y)
{
    const int tid = threadIdx.x;
    const long row = blockIdx.x;
    const float* xr = x + row * DIM;
    float v0 = xr[tid], v1 = xr[tid + 256];

    __shared__ float red[256];
    red[tid] = v0 + v1;
    __syncthreads();
    for (int st = 128; st > 0; st >>= 1) {
        if (tid < st) red[tid] += red[tid + st];
        __syncthreads();
    }
    float mean = red[0] * (1.0f / DIM);
    __syncthreads();
    float d0 = v0 - mean, d1 = v1 - mean;
    red[tid] = d0 * d0 + d1 * d1;
    __syncthreads();
    for (int st = 128; st > 0; st >>= 1) {
        if (tid < st) red[tid] += red[tid + st];
        __syncthreads();
    }
    float rstd = rsqrtf(red[0] * (1.0f / DIM) + 1e-5f);
    y[row * DIM + tid]       = d0 * rstd * g[tid]       + b[tid];
    y[row * DIM + tid + 256] = d1 * rstd * g[tid + 256] + b[tid + 256];
}

// ---------------- RoPE on q and k (first 32 dims of each head) ----------------
__global__ void rope_kernel(float* __restrict__ q, float* __restrict__ k)
{
    int idx = blockIdx.x * blockDim.x + threadIdx.x;   // T*HEADS*16
    if (idx >= TT * HEADS * 16) return;
    int i = idx & 15;
    int th = idx >> 4;
    int h = th % HEADS;
    int t = th / HEADS;
    float inv = __powf(10000.0f, -(float)i / 16.0f);
    float fr = (float)t * inv;
    float c = __cosf(fr), s = __sinf(fr);
    long base = (long)t * INNER + (long)h * DH;
    {
        float x1 = q[base + i], x2 = q[base + 16 + i];
        q[base + i]      = x1 * c - x2 * s;
        q[base + 16 + i] = x1 * s + x2 * c;
    }
    {
        float x1 = k[base + i], x2 = k[base + 16 + i];
        k[base + i]      = x1 * c - x2 * s;
        k[base + 16 + i] = x1 * s + x2 * c;
    }
}

// ---------------- head mixing (theta @ heads) + alibi + causal mask ----------------
__global__ void headmix_kernel(const float* __restrict__ src, const float* __restrict__ theta,
                               float* __restrict__ dst, int addAlibi, int causal)
{
    int ij = blockIdx.x * blockDim.x + threadIdx.x;
    if (ij >= TT * TT) return;
    int i = ij >> 9;           // /512
    int j = ij & 511;
    float vals[HEADS];
    #pragma unroll
    for (int h = 0; h < HEADS; ++h) vals[h] = src[(long)h * TT * TT + ij];
    float dist = -fabsf((float)(i - j));
    #pragma unroll
    for (int g = 0; g < HEADS; ++g) {
        float s = 0.f;
        #pragma unroll
        for (int h = 0; h < HEADS; ++h) s += theta[g * HEADS + h] * vals[h];
        if (addAlibi && g < ALIBI_HEADS) s += exp2f(-2.0f * (float)(g + 1)) * dist;
        if (causal && j > i) s = NEG_INF;
        dst[(long)g * TT * TT + ij] = s;
    }
}

// ---------------- softmax over rows of 512 (in place) ----------------
__global__ __launch_bounds__(256)
void softmax_kernel(float* __restrict__ p)
{
    const int tid = threadIdx.x;
    float* r = p + (long)blockIdx.x * TT;
    float v0 = r[tid], v1 = r[tid + 256];

    __shared__ float red[256];
    red[tid] = fmaxf(v0, v1);
    __syncthreads();
    for (int st = 128; st > 0; st >>= 1) {
        if (tid < st) red[tid] = fmaxf(red[tid], red[tid + st]);
        __syncthreads();
    }
    float mx = red[0];
    __syncthreads();
    float e0 = __expf(v0 - mx), e1 = __expf(v1 - mx);
    red[tid] = e0 + e1;
    __syncthreads();
    for (int st = 128; st > 0; st >>= 1) {
        if (tid < st) red[tid] += red[tid + st];
        __syncthreads();
    }
    float inv = 1.0f / red[0];
    r[tid]       = e0 * inv;
    r[tid + 256] = e1 * inv;
}

// ---------------- GEGLU: act = a * gelu_exact(g) ----------------
__global__ void geglu_kernel(const float* __restrict__ u, float* __restrict__ act)
{
    int idx = blockIdx.x * blockDim.x + threadIdx.x;   // T*FF_INNER
    if (idx >= TT * FF_INNER) return;
    int t = idx / FF_INNER;
    int f = idx - t * FF_INNER;
    const float* ur = u + (long)t * (2 * FF_INNER);
    float a = ur[f];
    float g = ur[FF_INNER + f];
    float gl = 0.5f * g * (1.0f + erff(g * 0.70710678118654752f));
    act[idx] = a * gl;
}

// ---------------- host side ----------------
static inline void gemm(hipStream_t s, const float* A, const float* B,
                        const float* bias, const float* resid, float* C,
                        int M, int N, int K,
                        long As0, long As1, long Bs0, long Bs1, long Cs0, long Cs1,
                        long batA, long batB, long batC, int batch, float scale)
{
    dim3 grid((N + BN - 1) / BN, (M + BM - 1) / BM, batch);
    const bool kal = (K % BK) == 0;
    const bool nal = (N % BN) == 0;
    const bool ak1 = (As1 == 1) && ((As0 & 3) == 0) && ((batA & 3) == 0);
    const bool bn1 = (Bs1 == 1) && ((Bs0 & 3) == 0) && ((batB & 3) == 0);

    if (kal && nal && ak1 && bn1)
        wmma_gemm_kernel<false, false, true, true><<<grid, 256, 0, s>>>(
            A, B, bias, resid, C, M, N, K, As0, As1, Bs0, Bs1, Cs0, Cs1, batA, batB, batC, scale);
    else if (kal && nal && ak1 && !bn1)
        wmma_gemm_kernel<false, false, true, false><<<grid, 256, 0, s>>>(
            A, B, bias, resid, C, M, N, K, As0, As1, Bs0, Bs1, Cs0, Cs1, batA, batB, batC, scale);
    else if (kal && !nal && ak1 && bn1)
        wmma_gemm_kernel<false, true, true, true><<<grid, 256, 0, s>>>(
            A, B, bias, resid, C, M, N, K, As0, As1, Bs0, Bs1, Cs0, Cs1, batA, batB, batC, scale);
    else
        wmma_gemm_kernel<true, true, false, false><<<grid, 256, 0, s>>>(
            A, B, bias, resid, C, M, N, K, As0, As1, Bs0, Bs1, Cs0, Cs1, batA, batB, batC, scale);
}

extern "C" void kernel_launch(void* const* d_in, const int* in_sizes, int n_in,
                              void* d_out, int out_size, void* d_ws, size_t ws_size,
                              hipStream_t stream)
{
    (void)in_sizes; (void)n_in; (void)out_size; (void)ws_size;

    const float* x     = (const float*)d_in[0];   // (1, 2, 2049, 512)
    const float* in_w  = (const float*)d_in[1];   // (4098, 512)
    const float* in_b  = (const float*)d_in[2];   // (512,)
    const float* out_w = (const float*)d_in[37];  // (512, 8196)
    const float* out_b = (const float*)d_in[38];  // (8196,)
    float* out = (float*)d_out;                   // (1, 4, 2049, 512)

    // workspace layout (f32)
    float* ws = (float*)d_ws;
    float* hA    = ws;
    float* hB    = hA   + (long)TT * DIM;
    float* y     = hB   + (long)TT * DIM;
    float* q     = y    + (long)TT * DIM;
    float* k     = q    + (long)TT * INNER;
    float* v     = k    + (long)TT * INNER;
    float* o     = v    + (long)TT * INNER;
    float* dots  = o    + (long)TT * INNER;
    float* dots2 = dots  + (long)HEADS * TT * TT;
    float* u     = dots2 + (long)HEADS * TT * TT;
    float* act   = u    + (long)TT * (2 * FF_INNER);

    auto prm = [&](int base, int i) { return (const float*)d_in[base + i]; };

    // input projection: hA[t][d] = x^T[t][cf] @ in_w + in_b
    // A element (t, i) = x[i*T + t]  (column-major view of x reshaped [CF][T])
    gemm(stream, x, in_w, in_b, nullptr, hA,
         TT, DIM, CF,
         /*A*/ 1, TT, /*B*/ DIM, 1, /*C*/ DIM, 1,
         0, 0, 0, 1, 1.0f);

    const int nblk_ij = (TT * TT + 255) / 256;
    const int nblk_rope = (TT * HEADS * 16 + 255) / 256;
    const int nblk_geglu = (TT * FF_INNER + 255) / 256;

    for (int stack = 0; stack < 2; ++stack) {
        const int base = (stack == 0) ? 3 : 20;
        const int causal = (stack == 0) ? 0 : 1;

        for (int l = 0; l < DEPTH; ++l) {
            const float* ln1_g = prm(base, 0) + (long)l * DIM;
            const float* ln1_b = prm(base, 1) + (long)l * DIM;
            const float* wq    = prm(base, 2) + (long)l * DIM * INNER;
            const float* wk    = prm(base, 3) + (long)l * DIM * INNER;
            const float* wv    = prm(base, 4) + (long)l * DIM * INNER;
            const float* wo    = prm(base, 5) + (long)l * INNER * DIM;
            const float* bo    = prm(base, 6) + (long)l * DIM;
            const float* preth = prm(base, 7) + (long)l * HEADS * HEADS;
            const float* posth = prm(base, 8) + (long)l * HEADS * HEADS;
            const float* ln2_g = prm(base, 9) + (long)l * DIM;
            const float* ln2_b = prm(base, 10) + (long)l * DIM;
            const float* wff1  = prm(base, 11) + (long)l * DIM * 2 * FF_INNER;
            const float* bff1  = prm(base, 12) + (long)l * 2 * FF_INNER;
            const float* wff2  = prm(base, 13) + (long)l * FF_INNER * DIM;
            const float* bff2  = prm(base, 14) + (long)l * DIM;

            // 1) y = LN(hA)
            layernorm_kernel<<<TT, 256, 0, stream>>>(hA, ln1_g, ln1_b, y);

            // 2) q,k,v = y @ wq/wk/wv
            gemm(stream, y, wq, nullptr, nullptr, q, TT, INNER, DIM,
                 DIM, 1, INNER, 1, INNER, 1, 0, 0, 0, 1, 1.0f);
            gemm(stream, y, wk, nullptr, nullptr, k, TT, INNER, DIM,
                 DIM, 1, INNER, 1, INNER, 1, 0, 0, 0, 1, 1.0f);
            gemm(stream, y, wv, nullptr, nullptr, v, TT, INNER, DIM,
                 DIM, 1, INNER, 1, INNER, 1, 0, 0, 0, 1, 1.0f);

            // 3) rope on q,k
            rope_kernel<<<nblk_rope, 256, 0, stream>>>(q, k);

            // 4) dots[h] = SCALE * q_h @ k_h^T   (batched over heads)
            gemm(stream, q, k, nullptr, nullptr, dots, TT, TT, DH,
                 /*A*/ INNER, 1, /*B (k^T)*/ 1, INNER, /*C*/ TT, 1,
                 /*bat*/ DH, DH, (long)TT * TT, HEADS, SCALE);

            // 5) dots2 = pre_th-mix(dots) + alibi (+ causal mask)
            headmix_kernel<<<nblk_ij, 256, 0, stream>>>(dots, preth, dots2, 1, causal);

            // 6) softmax rows
            softmax_kernel<<<HEADS * TT, 256, 0, stream>>>(dots2);

            // 7) dots = post_th-mix(attn)
            headmix_kernel<<<nblk_ij, 256, 0, stream>>>(dots2, posth, dots, 0, 0);

            // 8) o[.,h] = attn_h @ v_h
            gemm(stream, dots, v, nullptr, nullptr, o, TT, DH, TT,
                 /*A*/ TT, 1, /*B*/ INNER, 1, /*C*/ INNER, 1,
                 /*bat*/ (long)TT * TT, DH, DH, HEADS, 1.0f);

            // 9) hB = hA + o @ wo + bo
            gemm(stream, o, wo, bo, hA, hB, TT, DIM, INNER,
                 INNER, 1, DIM, 1, DIM, 1, 0, 0, 0, 1, 1.0f);

            // 10) y = LN(hB)
            layernorm_kernel<<<TT, 256, 0, stream>>>(hB, ln2_g, ln2_b, y);

            // 11) u = y @ w_ff1 + b_ff1
            gemm(stream, y, wff1, bff1, nullptr, u, TT, 2 * FF_INNER, DIM,
                 DIM, 1, 2 * FF_INNER, 1, 2 * FF_INNER, 1, 0, 0, 0, 1, 1.0f);

            // 12) act = a * gelu(g)
            geglu_kernel<<<nblk_geglu, 256, 0, stream>>>(u, act);

            // 13) hA = hB + act @ w_ff2 + b_ff2
            gemm(stream, act, wff2, bff2, hB, hA, TT, DIM, FF_INNER,
                 FF_INNER, 1, DIM, 1, DIM, 1, 0, 0, 0, 1, 1.0f);
        }

        // final LN of this stack (in place: each thread holds its values in regs)
        layernorm_kernel<<<TT, 256, 0, stream>>>(hA, prm(base, 15), prm(base, 16), hA);
    }

    // output projection, stored transposed: out[n*T + t] = (hA @ out_w + out_b)[t][n]
    gemm(stream, hA, out_w, out_b, nullptr, out, TT, NOUT, DIM,
         DIM, 1, NOUT, 1, /*C*/ 1, TT, 0, 0, 0, 1, 1.0f);
}
